// RopelessMLA_31018253812046
// MI455X (gfx1250) — matrix operand order repaired and checked by reading, not compiled, not run on hardware
//
#include <hip/hip_runtime.h>
#include <stdint.h>

// ---------------- problem constants ----------------
#define DM    1024          // d_model
#define NH    16            // heads
#define PROJ  256           // q/kv low-rank dim
#define DHD   64            // head dim
#define SEQ   2048
#define BAT   2
#define NROW  (BAT*SEQ)     // 4096 tokens
#define LN_EPS 1e-5f
#define LOG2E  1.44269504088896340736f

typedef __attribute__((ext_vector_type(16))) __bf16 v16bf;
typedef __attribute__((ext_vector_type(8)))  float  v8f;
typedef __attribute__((ext_vector_type(4)))  uint32_t v4u;
typedef __attribute__((ext_vector_type(8)))  uint32_t v8u;

union Frag { v16bf v; uint4 u[2]; uint32_t w[8]; };

// fp32 -> bf16 (round to nearest even)
__device__ __forceinline__ uint32_t f2bf(float f) {
    uint32_t x = __float_as_uint(f);
    return (x + 0x7fffu + ((x >> 16) & 1u)) >> 16;
}
__device__ __forceinline__ uint32_t pack2(float lo, float hi) {
    return f2bf(lo) | (f2bf(hi) << 16);
}

// A fragment (16x32 bf16): lane = row(M), holds K = kb+half*16+0..15 (contiguous)
__device__ __forceinline__ v16bf afrag(const uint16_t* p) {
    Frag f; f.u[0] = *(const uint4*)p; f.u[1] = *(const uint4*)(p + 8); return f.v;
}
// B fragment (32x16 bf16): lane = col(N), holds K = kb+half*16+0..15 (contiguous)
__device__ __forceinline__ v16bf bfrag(const uint16_t* p) {
    Frag f; f.u[0] = *(const uint4*)p; f.u[1] = *(const uint4*)(p + 8); return f.v;
}
__device__ __forceinline__ v8f wmma_bf16(v16bf a, v16bf b, v8f c) {
    return __builtin_amdgcn_wmma_f32_16x16x32_bf16(false, a, false, b, (short)0, c, false, false);
}

// -------- CDNA5 async global->LDS copy (ASYNCcnt) --------
// LDS byte address = low 32 bits of the generic pointer (FLAT aperture rule).
__device__ __forceinline__ void async_ld_b128(void* lds_dst, const void* gsrc) {
    uint32_t l = (uint32_t)(uintptr_t)lds_dst;
    asm volatile("global_load_async_to_lds_b128 %0, %1, off"
                 :: "v"(l), "v"(gsrc) : "memory");
}
__device__ __forceinline__ void wait_async4() {
    asm volatile("s_wait_asynccnt 0x4" ::: "memory");
}
__device__ __forceinline__ void wait_async0() {
    asm volatile("s_wait_asynccnt 0x0" ::: "memory");
}

// -------- CDNA5 Tensor Data Mover: 32(K) x 256(N) bf16 weight tile -> padded LDS [256][40]
// D# per ISA ch.8: group0 {count=1, lds_addr, global_addr, type=2};
// group1 {data_size=1(2B), pad_enable, pad_interval=3 (16 DWORDs), pad_amount=3 (4 DWORDs),
//         tensor_dim0=32, tensor_dim1=256, tile_dim0=32, tile_dim1=256, dim0_stride=ldk}.
// Padding inserts 16B after each 64B row => exactly the [256][40] bank-padded layout.
__device__ __forceinline__ void tdm_load_b_tile(const uint16_t* gsrc, uint32_t ldk,
                                                void* lds_dst) {
    uint64_t ga = (uint64_t)(uintptr_t)gsrc;
    uint32_t la = (uint32_t)(uintptr_t)lds_dst;
    v4u g0; v8u g1; v4u gz;
    g0[0] = 1u;                                   // count=1 (valid user descriptor)
    g0[1] = la;                                   // lds_addr
    g0[2] = (uint32_t)ga;                         // global_addr[31:0]
    g0[3] = (uint32_t)(ga >> 32) | 0x80000000u;   // global_addr[56:32] | type=2
    g1[0] = (1u << 16) | (1u << 20) | (3u << 22) | (3u << 25);
    g1[1] = 32u << 16;                            // tensor_dim0 = 32 (lo 16)
    g1[2] = 256u << 16;                           // tensor_dim0 hi | tensor_dim1 lo = 256
    g1[3] = 32u << 16;                            // tensor_dim1 hi | tile_dim0 = 32
    g1[4] = 256u;                                 // tile_dim1 = 256, tile_dim2 = 0
    g1[5] = ldk;                                  // tensor_dim0_stride (elements)
    g1[6] = 0u;
    g1[7] = 0u;                                   // tensor_dim1_stride unused (2D)
    gz[0] = 0u; gz[1] = 0u; gz[2] = 0u; gz[3] = 0u;
    asm volatile("tensor_load_to_lds %0, %1, %2, %3"
                 :: "s"(g0), "s"(g1), "s"(gz), "s"(gz) : "memory");
}
__device__ __forceinline__ void wait_tensor0() { __builtin_amdgcn_s_wait_tensorcnt(0); }
__device__ __forceinline__ void wait_tensor1() { __builtin_amdgcn_s_wait_tensorcnt(1); }

// ---------------- small conversion kernels ----------------
__global__ __launch_bounds__(256) void cvt_bf16_kernel(const float* __restrict__ in,
                                                       uint16_t* __restrict__ out, int n) {
    int i = blockIdx.x * 256 + threadIdx.x;
    if (i < n) out[i] = (uint16_t)f2bf(in[i]);
}

// in: f32 [R][C]  ->  out: bf16 [C][R]
__global__ __launch_bounds__(256) void transpose_bf16_kernel(const float* __restrict__ in,
                                                             uint16_t* __restrict__ out,
                                                             int R, int C) {
    __shared__ uint16_t tile[32][33];
    int bx = blockIdx.x * 32;
    int by = blockIdx.y * 32;
    int tx = threadIdx.x & 31;
    int ty = threadIdx.x >> 5;
    #pragma unroll
    for (int j = 0; j < 32; j += 8)
        tile[ty + j][tx] = (uint16_t)f2bf(in[(by + ty + j) * C + bx + tx]);
    __syncthreads();
    #pragma unroll
    for (int j = 0; j < 32; j += 8)
        out[(bx + ty + j) * R + by + tx] = tile[tx][ty + j];
}

// ---------------- kernel A: down-projection + LayerNorm ----------------
__global__ __launch_bounds__(256) void proj_ln_kernel(
        const uint16_t* __restrict__ xb,
        const uint16_t* __restrict__ wdqt, const uint16_t* __restrict__ wdkvt,
        const float* __restrict__ qg, const float* __restrict__ qb,
        const float* __restrict__ kg, const float* __restrict__ kb2,
        uint16_t* __restrict__ cq, uint16_t* __restrict__ ckv) {
    __shared__ uint16_t Bl[2][256][40];
    const int path = blockIdx.y;
    const uint16_t* wt = path ? wdkvt : wdqt;       // [256][1024] bf16 (N-major)
    const float* g  = path ? kg  : qg;
    const float* bb = path ? kb2 : qb;
    uint16_t* outp  = path ? ckv : cq;              // [4096][256] bf16

    const int tid  = threadIdx.x;
    const int lane = tid & 31;
    const int wave = tid >> 5;
    const int l15  = lane & 15;
    const int hi   = lane >> 4;
    const int rbase = blockIdx.x * 128 + wave * 16;
    const int arow  = rbase + l15;

    v8f z = {};
    v8f acc[16];
    #pragma unroll
    for (int t = 0; t < 16; t++) acc[t] = z;

    if (wave == 0) tdm_load_b_tile(wt, DM, &Bl[0][0][0]);
    v16bf a = afrag(xb + (size_t)arow * DM + hi * 16);   // k-step 0, in flight early
    int cur = 0;
    for (int kb = 0; kb < DM; kb += 32) {
        if (wave == 0) {
            if (kb + 32 < DM) { tdm_load_b_tile(wt + kb + 32, DM, &Bl[cur ^ 1][0][0]); wait_tensor1(); }
            else              { wait_tensor0(); }
        }
        __syncthreads();
        // prefetch next A fragment (global) behind this k-step's WMMAs
        v16bf anext = a;
        if (kb + 32 < DM) anext = afrag(xb + (size_t)arow * DM + kb + 32 + hi * 16);
        // software-pipelined B fragments: tile t+1 loads while tile t multiplies
        v16bf bcur = bfrag(&Bl[cur][l15][hi * 16]);
        #pragma unroll
        for (int t = 0; t < 16; t++) {
            v16bf bnext = bcur;
            if (t < 15) bnext = bfrag(&Bl[cur][(t + 1) * 16 + l15][hi * 16]);
            acc[t] = wmma_bf16(a, bcur, acc[t]);
            bcur = bnext;
        }
        __syncthreads();
        a = anext;
        cur ^= 1;
    }

    float gc[16], bc[16];
    #pragma unroll
    for (int t = 0; t < 16; t++) { gc[t] = g[t * 16 + l15]; bc[t] = bb[t * 16 + l15]; }

    // LayerNorm over the 256 columns of each row, then store bf16
    #pragma unroll
    for (int i = 0; i < 8; i++) {
        float s = 0.f, ss = 0.f;
        #pragma unroll
        for (int t = 0; t < 16; t++) { float v = acc[t][i]; s += v; ss += v * v; }
        #pragma unroll
        for (int m = 8; m >= 1; m >>= 1) { s += __shfl_xor(s, m, 32); ss += __shfl_xor(ss, m, 32); }
        float mu  = s * (1.0f / 256.0f);
        float var = ss * (1.0f / 256.0f) - mu * mu;
        float rs  = rsqrtf(var + LN_EPS);
        int row = rbase + i + (hi ? 8 : 0);
        #pragma unroll
        for (int t = 0; t < 16; t++) {
            float y = (acc[t][i] - mu) * rs * gc[t] + bc[t];
            outp[row * PROJ + t * 16 + l15] = (uint16_t)f2bf(y);
        }
    }
}

// ---------------- kernel B: up-projections -> Q / K / V^T ----------------
__global__ __launch_bounds__(256) void up_proj_kernel(
        const uint16_t* __restrict__ cq,  const uint16_t* __restrict__ ckv,
        const uint16_t* __restrict__ wuqt, const uint16_t* __restrict__ wukvt,
        uint16_t* __restrict__ Qb, uint16_t* __restrict__ Kb, uint16_t* __restrict__ Vt) {
    __shared__ uint16_t Bl[2][256][40];
    const int y    = blockIdx.y;
    const bool isQ = (y < 4);
    const uint16_t* A  = isQ ? cq   : ckv;     // [4096][256]
    const uint16_t* Bt = isQ ? wuqt : wukvt;   // [N][256]
    const int cbase = (isQ ? y : y - 4) * 256;
    const uint16_t* Bslab = Bt + (size_t)cbase * PROJ;

    const int tid  = threadIdx.x;
    const int lane = tid & 31;
    const int wave = tid >> 5;
    const int l15  = lane & 15;
    const int hi   = lane >> 4;
    const int rbase = blockIdx.x * 128 + wave * 16;
    const int arow  = rbase + l15;

    v8f z = {};
    v8f acc[16];
    #pragma unroll
    for (int t = 0; t < 16; t++) acc[t] = z;

    if (wave == 0) tdm_load_b_tile(Bslab, PROJ, &Bl[0][0][0]);
    v16bf a = afrag(A + (size_t)arow * PROJ + hi * 16);
    int cur = 0;
    for (int kb = 0; kb < PROJ; kb += 32) {
        if (wave == 0) {
            if (kb + 32 < PROJ) { tdm_load_b_tile(Bslab + kb + 32, PROJ, &Bl[cur ^ 1][0][0]); wait_tensor1(); }
            else                { wait_tensor0(); }
        }
        __syncthreads();
        v16bf anext = a;
        if (kb + 32 < PROJ) anext = afrag(A + (size_t)arow * PROJ + kb + 32 + hi * 16);
        v16bf bcur = bfrag(&Bl[cur][l15][hi * 16]);
        #pragma unroll
        for (int t = 0; t < 16; t++) {
            v16bf bnext = bcur;
            if (t < 15) bnext = bfrag(&Bl[cur][(t + 1) * 16 + l15][hi * 16]);
            acc[t] = wmma_bf16(a, bcur, acc[t]);
            bcur = bnext;
        }
        __syncthreads();
        a = anext;
        cur ^= 1;
    }

    if (isQ) {
        #pragma unroll
        for (int t = 0; t < 16; t++) {
            int j = cbase + t * 16 + l15, h = j >> 6, dh = j & 63;
            #pragma unroll
            for (int i = 0; i < 8; i++) {
                int tok = rbase + i + (hi ? 8 : 0);
                int b = tok >> 11, s = tok & 2047;
                Qb[(((b * NH + h) * SEQ) + s) * DHD + dh] = (uint16_t)f2bf(acc[t][i] * 0.125f);
            }
        }
    } else if (cbase < DM) {            // K columns
        #pragma unroll
        for (int t = 0; t < 16; t++) {
            int j = cbase + t * 16 + l15, h = j >> 6, dh = j & 63;
            #pragma unroll
            for (int i = 0; i < 8; i++) {
                int tok = rbase + i + (hi ? 8 : 0);
                int b = tok >> 11, s = tok & 2047;
                Kb[(((b * NH + h) * SEQ) + s) * DHD + dh] = (uint16_t)f2bf(acc[t][i]);
            }
        }
    } else {                            // V columns -> transposed, packed 128-bit stores
        int b  = rbase >> 11;
        int sb = (rbase & 2047) + (hi ? 8 : 0);
        #pragma unroll
        for (int t = 0; t < 16; t++) {
            int j = cbase + t * 16 + l15 - DM, h = j >> 6, dh = j & 63;
            uint4 u;
            u.x = pack2(acc[t][0], acc[t][1]);
            u.y = pack2(acc[t][2], acc[t][3]);
            u.z = pack2(acc[t][4], acc[t][5]);
            u.w = pack2(acc[t][6], acc[t][7]);
            *(uint4*)(Vt + ((b * NH + h) * DHD + dh) * SEQ + sb) = u;
        }
    }
}

// ---------------- kernel C: flash-style causal attention ----------------
// grid = (S/128, NH, B); 8 waves; wave owns 16 queries x DH=64 output.
// K/V tiles (64 keys) staged via per-lane async-to-LDS, double buffered.
__global__ __launch_bounds__(256) void attn_kernel(
        const uint16_t* __restrict__ Q, const uint16_t* __restrict__ K,
        const uint16_t* __restrict__ Vt, uint16_t* __restrict__ O) {
    __shared__ uint16_t Kl[2][64][72];
    __shared__ uint16_t Vl[2][64][72];

    const int tid  = threadIdx.x;
    const int lane = tid & 31;
    const int wave = tid >> 5;
    const int l15  = lane & 15;
    const int hi   = lane >> 4;
    const int qblk = blockIdx.x, h = blockIdx.y, b = blockIdx.z;
    const int qbase = qblk * 128 + wave * 16;

    const uint16_t* Qbh = Q  + (size_t)(b * NH + h) * SEQ * DHD;
    const uint16_t* Kbh = K  + (size_t)(b * NH + h) * SEQ * DHD;
    const uint16_t* Vbh = Vt + (size_t)(b * NH + h) * DHD * SEQ;

    // Q^T B-fragments (d = 0..31, 32..63), queries fixed for the wave
    v16bf qf0 = bfrag(Qbh + (qbase + l15) * DHD + 0  + hi * 16);
    v16bf qf1 = bfrag(Qbh + (qbase + l15) * DHD + 32 + hi * 16);

    v8f z = {};
    v8f ot[4]; ot[0] = z; ot[1] = z; ot[2] = z; ot[3] = z;
    float mrow = -INFINITY, lrow = 0.f;
    const int q = qbase + l15;
    const int kend = qblk * 128 + 128;

    // stage first 64-key block: K rows + V^T rows, 4 async b128 per thread
    {
        #pragma unroll
        for (int j = 0; j < 2; j++) {
            int c = tid + 256 * j, row = c >> 3, col = c & 7;
            async_ld_b128(&Kl[0][row][col * 8], Kbh + (size_t)row * DHD + col * 8);
        }
        #pragma unroll
        for (int j = 0; j < 2; j++) {
            int c = tid + 256 * j, row = c >> 3, col = c & 7;
            async_ld_b128(&Vl[0][row][col * 8], Vbh + (size_t)row * SEQ + col * 8);
        }
    }

    int cur = 0;
    for (int kblk = 0; kblk < kend; kblk += 64) {
        if (kblk + 64 < kend) {
            #pragma unroll
            for (int j = 0; j < 2; j++) {
                int c = tid + 256 * j, row = c >> 3, col = c & 7;
                async_ld_b128(&Kl[cur ^ 1][row][col * 8],
                              Kbh + (size_t)(kblk + 64 + row) * DHD + col * 8);
            }
            #pragma unroll
            for (int j = 0; j < 2; j++) {
                int c = tid + 256 * j, row = c >> 3, col = c & 7;
                async_ld_b128(&Vl[cur ^ 1][row][col * 8],
                              Vbh + (size_t)row * SEQ + kblk + 64 + col * 8);
            }
            wait_async4();
        } else {
            wait_async0();
        }
        __syncthreads();

        // scores^T : four 16-key tiles over 64 keys (K-fragments software-pipelined)
        v8f st[4]; st[0] = z; st[1] = z; st[2] = z; st[3] = z;
        v16bf k0 = afrag(&Kl[cur][l15][0  + hi * 16]);
        v16bf k1 = afrag(&Kl[cur][l15][32 + hi * 16]);
        #pragma unroll
        for (int kt = 0; kt < 4; kt++) {
            v16bf n0 = k0, n1 = k1;
            if (kt < 3) {
                n0 = afrag(&Kl[cur][(kt + 1) * 16 + l15][0  + hi * 16]);
                n1 = afrag(&Kl[cur][(kt + 1) * 16 + l15][32 + hi * 16]);
            }
            st[kt] = wmma_bf16(k0, qf0, st[kt]);
            st[kt] = wmma_bf16(k1, qf1, st[kt]);
            k0 = n0; k1 = n1;
        }

        // mask (tril + zero->-inf quirk) and local max
        float p[4][8];
        float mnew = mrow;
        #pragma unroll
        for (int kt = 0; kt < 4; kt++)
            #pragma unroll
            for (int i = 0; i < 8; i++) {
                int key = kblk + kt * 16 + i + (hi ? 8 : 0);
                float s = st[kt][i];
                s = (key <= q && s != 0.0f) ? s : -INFINITY;
                p[kt][i] = s;
                mnew = fmaxf(mnew, s);
            }
        mnew = fmaxf(mnew, __shfl_xor(mnew, 16, 32));   // merge key halves

        float alpha = (mnew == -INFINITY) ? 1.0f : exp2f((mrow - mnew) * LOG2E);
        float ls = 0.f;
        #pragma unroll
        for (int kt = 0; kt < 4; kt++)
            #pragma unroll
            for (int i = 0; i < 8; i++) {
                float e = (p[kt][i] == -INFINITY) ? 0.f : exp2f((p[kt][i] - mnew) * LOG2E);
                p[kt][i] = e; ls += e;
            }
        ls += __shfl_xor(ls, 16, 32);
        lrow = lrow * alpha + ls;
        mrow = mnew;
        #pragma unroll
        for (int dt = 0; dt < 4; dt++) ot[dt] = ot[dt] * alpha;

        // Build two P^T B-fragments (keys 0..31 and 32..63 of the block)
        Frag pf[2];
        #pragma unroll
        for (int g2 = 0; g2 < 2; g2++) {
            float* p0 = p[2 * g2];
            float* p1 = p[2 * g2 + 1];
            #pragma unroll
            for (int v2 = 0; v2 < 4; v2++) {
                float a0 = __shfl_xor(p1[2 * v2],     16, 32);
                float a1 = __shfl_xor(p1[2 * v2 + 1], 16, 32);
                float lo0 = hi ? a0 : p0[2 * v2];
                float lo1 = hi ? a1 : p0[2 * v2 + 1];
                pf[g2].w[v2] = pack2(lo0, lo1);
                float b0 = __shfl_xor(p0[2 * v2],     16, 32);
                float b1 = __shfl_xor(p0[2 * v2 + 1], 16, 32);
                float h0 = hi ? p1[2 * v2]     : b0;
                float h1 = hi ? p1[2 * v2 + 1] : b1;
                pf[g2].w[v2 + 4] = pack2(h0, h1);
            }
        }

        // out^T += V^T @ P^T  (A = V^T rows = dh; V-fragments software-pipelined)
        v16bf va = afrag(&Vl[cur][l15][0  + hi * 16]);
        v16bf vb = afrag(&Vl[cur][l15][32 + hi * 16]);
        #pragma unroll
        for (int dt = 0; dt < 4; dt++) {
            v16bf na = va, nb = vb;
            if (dt < 3) {
                na = afrag(&Vl[cur][(dt + 1) * 16 + l15][0  + hi * 16]);
                nb = afrag(&Vl[cur][(dt + 1) * 16 + l15][32 + hi * 16]);
            }
            ot[dt] = wmma_bf16(va, pf[0].v, ot[dt]);
            ot[dt] = wmma_bf16(vb, pf[1].v, ot[dt]);
            va = na; vb = nb;
        }
        __syncthreads();
        cur ^= 1;
    }

    // normalize and store O[token][d] (packed: 8 consecutive dh per lane)
    float rl = 1.0f / lrow;
    int token = b * SEQ + qbase + l15;
    #pragma unroll
    for (int dt = 0; dt < 4; dt++) {
        uint4 u;
        u.x = pack2(ot[dt][0] * rl, ot[dt][1] * rl);
        u.y = pack2(ot[dt][2] * rl, ot[dt][3] * rl);
        u.z = pack2(ot[dt][4] * rl, ot[dt][5] * rl);
        u.w = pack2(ot[dt][6] * rl, ot[dt][7] * rl);
        *(uint4*)(O + (size_t)token * DM + h * DHD + dt * 16 + (hi ? 8 : 0)) = u;
    }
}

// ---------------- kernel D: output projection  out = O @ W_o^T ----------------
__global__ __launch_bounds__(256) void out_proj_kernel(
        const uint16_t* __restrict__ O, const uint16_t* __restrict__ Wo,
        float* __restrict__ out) {
    __shared__ uint16_t Bl[2][256][40];
    const int tid  = threadIdx.x;
    const int lane = tid & 31;
    const int wave = tid >> 5;
    const int l15  = lane & 15;
    const int hi   = lane >> 4;
    const int rbase = blockIdx.x * 128 + wave * 16;
    const int cbase = blockIdx.y * 256;
    const int arow  = rbase + l15;
    const uint16_t* Wslab = Wo + (size_t)cbase * DM;

    v8f z = {};
    v8f acc[16];
    #pragma unroll
    for (int t = 0; t < 16; t++) acc[t] = z;

    if (wave == 0) tdm_load_b_tile(Wslab, DM, &Bl[0][0][0]);
    v16bf a = afrag(O + (size_t)arow * DM + hi * 16);
    int cur = 0;
    for (int kb = 0; kb < DM; kb += 32) {
        if (wave == 0) {
            if (kb + 32 < DM) { tdm_load_b_tile(Wslab + kb + 32, DM, &Bl[cur ^ 1][0][0]); wait_tensor1(); }
            else              { wait_tensor0(); }
        }
        __syncthreads();
        v16bf anext = a;
        if (kb + 32 < DM) anext = afrag(O + (size_t)arow * DM + kb + 32 + hi * 16);
        v16bf bcur = bfrag(&Bl[cur][l15][hi * 16]);
        #pragma unroll
        for (int t = 0; t < 16; t++) {
            v16bf bnext = bcur;
            if (t < 15) bnext = bfrag(&Bl[cur][(t + 1) * 16 + l15][hi * 16]);
            acc[t] = wmma_bf16(a, bcur, acc[t]);
            bcur = bnext;
        }
        __syncthreads();
        a = anext;
        cur ^= 1;
    }
    #pragma unroll
    for (int t = 0; t < 16; t++) {
        int j = cbase + t * 16 + l15;
        #pragma unroll
        for (int i = 0; i < 8; i++)
            out[(size_t)(rbase + i + (hi ? 8 : 0)) * DM + j] = acc[t][i];
    }
}

// ---------------- host side ----------------
extern "C" void kernel_launch(void* const* d_in, const int* in_sizes, int n_in,
                              void* d_out, int out_size, void* d_ws, size_t ws_size,
                              hipStream_t stream) {
    const float* x     = (const float*)d_in[0];
    const float* W_dq  = (const float*)d_in[1];
    const float* W_uq  = (const float*)d_in[2];
    const float* q_g   = (const float*)d_in[3];
    const float* q_b   = (const float*)d_in[4];
    const float* W_dkv = (const float*)d_in[5];
    const float* W_ukv = (const float*)d_in[6];
    const float* kv_g  = (const float*)d_in[7];
    const float* kv_b  = (const float*)d_in[8];
    const float* W_o   = (const float*)d_in[9];
    float* outp = (float*)d_out;

    // workspace layout (bf16 elements)
    uint16_t* ws = (uint16_t*)d_ws;
    size_t off = 0;
    uint16_t* xb    = ws + off; off += (size_t)NROW * DM;        // 4096x1024
    uint16_t* wdqt  = ws + off; off += (size_t)PROJ * DM;        // [256][1024]
    uint16_t* wdkvt = ws + off; off += (size_t)PROJ * DM;
    uint16_t* wuqt  = ws + off; off += (size_t)DM * PROJ;        // [1024][256]
    uint16_t* wukvt = ws + off; off += (size_t)(2 * DM) * PROJ;  // [2048][256]
    uint16_t* wob   = ws + off; off += (size_t)DM * DM;          // [1024][1024]
    uint16_t* cq    = ws + off; off += (size_t)NROW * PROJ;
    uint16_t* ckv   = ws + off; off += (size_t)NROW * PROJ;
    uint16_t* Qb    = ws + off; off += (size_t)NROW * DM;        // [B][H][S][64]
    uint16_t* Kb    = ws + off; off += (size_t)NROW * DM;
    uint16_t* Vt    = ws + off; off += (size_t)NROW * DM;        // [B][H][64][S]
    uint16_t* Ob    = ws + off; off += (size_t)NROW * DM;

    // 1. conversions / weight transposes
    cvt_bf16_kernel<<<(NROW * DM) / 256, 256, 0, stream>>>(x, xb, NROW * DM);
    transpose_bf16_kernel<<<dim3(PROJ / 32, DM / 32), 256, 0, stream>>>(W_dq,  wdqt,  DM, PROJ);
    transpose_bf16_kernel<<<dim3(PROJ / 32, DM / 32), 256, 0, stream>>>(W_dkv, wdkvt, DM, PROJ);
    transpose_bf16_kernel<<<dim3(DM / 32, PROJ / 32), 256, 0, stream>>>(W_uq,  wuqt,  PROJ, DM);
    transpose_bf16_kernel<<<dim3((2 * DM) / 32, PROJ / 32), 256, 0, stream>>>(W_ukv, wukvt, PROJ, 2 * DM);
    cvt_bf16_kernel<<<(DM * DM) / 256, 256, 0, stream>>>(W_o, wob, DM * DM);

    // 2. down-proj + LN (both paths)
    proj_ln_kernel<<<dim3(NROW / 128, 2), 256, 0, stream>>>(
        xb, wdqt, wdkvt, q_g, q_b, kv_g, kv_b, cq, ckv);

    // 3. up-proj -> Q (scaled), K, V^T
    up_proj_kernel<<<dim3(NROW / 128, 12), 256, 0, stream>>>(
        cq, ckv, wuqt, wukvt, Qb, Kb, Vt);

    // 4. causal attention (flash-style)
    attn_kernel<<<dim3(SEQ / 128, NH, BAT), 256, 0, stream>>>(Qb, Kb, Vt, Ob);

    // 5. output projection
    out_proj_kernel<<<dim3(NROW / 128, DM / 256), 256, 0, stream>>>(Ob, wob, outp);
}
